// GroupedQueryAttention_52974126629320
// MI455X (gfx1250) — matrix operand order repaired and checked by reading, not compiled, not run on hardware
//
#include <hip/hip_runtime.h>
#include <stdint.h>

// ---------------- problem constants ----------------
#define BATCH   2
#define S_LEN   2048
#define DMODEL  2048
#define NHEAD   32
#define NKV     8
#define HDIM    64
#define KV_D    (NKV * HDIM)     // 512
#define BS      (BATCH * S_LEN)  // 4096 rows

typedef __attribute__((ext_vector_type(16))) __bf16 v16bf;
typedef __attribute__((ext_vector_type(8)))  float  v8f;

// round-to-nearest-even fp32 -> bf16 (raw ushort storage)
static __device__ __forceinline__ unsigned short f32_to_bf16(float f) {
  unsigned int u = __float_as_uint(f);
  u += 0x7fffu + ((u >> 16) & 1u);
  return (unsigned short)(u >> 16);
}

// A-fragment (16x32, 16-bit): lane holds row M = lane&15.
// element i -> K = i + 16*(lane>>4): one contiguous 16-elem chunk.
static __device__ __forceinline__ v16bf load_a_frag(const unsigned short* base, int lhi) {
  v16bf a;
  ((uint4*)&a)[0] = *(const uint4*)(base + 16 * lhi);
  ((uint4*)&a)[1] = *(const uint4*)(base + 16 * lhi + 8);
  return a;
}
// B-fragment (32x16, 16-bit): lane holds col N = lane&15,
// element i -> K = i + 16*(lane>>4): one contiguous 16-elem (32B) chunk.
static __device__ __forceinline__ v16bf load_b_frag(const unsigned short* p) {
  v16bf b;
  ((uint4*)&b)[0] = *(const uint4*)(p);
  ((uint4*)&b)[1] = *(const uint4*)(p + 8);
  return b;
}

// gfx1250 async Global->LDS copy, 16 bytes per lane (GV addressing).
// VDST = 32-bit LDS byte offset (low 32 bits of the generic shared pointer,
// per the ISA aperture rule); VADDR = 64-bit global address.
static __device__ __forceinline__ void async_copy_b128(unsigned lds_off,
                                                       const void* gptr) {
  unsigned long long ga = (unsigned long long)(uintptr_t)gptr;
  asm volatile("global_load_async_to_lds_b128 %0, %1, off"
               :: "v"(lds_off), "v"(ga)
               : "memory");
}
static __device__ __forceinline__ void wait_asynccnt0() {
  asm volatile("s_wait_asynccnt 0x0" ::: "memory");
}

// ---------------- elementwise fp32 -> bf16 ----------------
__global__ void cvt_bf16_kernel(const float* __restrict__ in,
                                unsigned short* __restrict__ out, size_t n) {
  size_t i = (size_t)blockIdx.x * blockDim.x + threadIdx.x;
  if (i < n) out[i] = f32_to_bf16(in[i]);
}

// ---------------- RoPE + convert: fp32 (BS x width) -> bf16 ----------------
__global__ void rope_cvt_kernel(const float* __restrict__ X,
                                const float* __restrict__ cs,
                                const float* __restrict__ sn,
                                unsigned short* __restrict__ O, int width) {
  size_t idx = (size_t)blockIdx.x * blockDim.x + threadIdx.x;
  size_t total = (size_t)BS * width;
  if (idx >= total) return;
  int    col = (int)(idx % width);
  size_t row = idx / width;
  int s = (int)(row & (S_LEN - 1));
  int d = col & (HDIM - 1);
  float v = X[idx];
  float other = (d < 32) ? -X[idx + 32] : X[idx - 32];
  float r = v * cs[s * HDIM + d] + other * sn[s * HDIM + d];
  O[idx] = f32_to_bf16(r);
}

// ---------------- V transpose: fp32 (BS x 512) -> bf16 (B*G*64, S) ----------------
__global__ void vt_cvt_kernel(const float* __restrict__ Vf,
                              unsigned short* __restrict__ Vt) {
  size_t idx = (size_t)blockIdx.x * blockDim.x + threadIdx.x;
  size_t total = (size_t)BS * KV_D;
  if (idx >= total) return;
  int    col = (int)(idx % KV_D);
  size_t row = idx / KV_D;
  int b = (int)(row / S_LEN);
  int t = (int)(row & (S_LEN - 1));
  int g = col >> 6;
  int d = col & 63;
  Vt[((size_t)(b * NKV + g) * HDIM + d) * S_LEN + t] = f32_to_bf16(Vf[idx]);
}

// ---------------- WMMA GEMM: C(MxN) = A(MxK) * W(NxK)^T ----------------
// block tile 128x128, 8 waves: 4 along M (32 rows) x 2 along N (64 cols).
// A/B K-tiles (128x32 bf16 = 8KB each) double-buffered in LDS, filled with
// gfx1250 async Global->LDS copies overlapped with the 8 WMMAs per K-step.
__global__ __launch_bounds__(256)
void gemm_bf16_nt(int M, int N, int K,
                  const unsigned short* __restrict__ A, int lda,
                  const unsigned short* __restrict__ W, int ldw,
                  float* __restrict__ C, int ldc) {
  __shared__ unsigned short sA[2][128][32];
  __shared__ unsigned short sB[2][128][32];

  const int tid  = threadIdx.x;
  const int lane = tid & 31;
  const int wv   = tid >> 5;
  const int lhi  = lane >> 4;
  const int llo  = lane & 15;
  const int wm   = wv & 3;
  const int wn   = wv >> 2;
  const int m_blk = blockIdx.y * 128;
  const int n_blk = blockIdx.x * 128;

  // per-thread copy coordinates: 8KB tile = 512 x 16B chunks, 2 per thread
  const int c_row0 = (tid * 8) >> 5;          // chunk 0 row  (tid*8 elems /32)
  const int c_col0 = (tid * 8) & 31;          // chunk 0 col (elements)
  const int c_row1 = ((tid + 256) * 8) >> 5;  // chunk 1
  const int c_col1 = ((tid + 256) * 8) & 31;

  auto stage = [&](int buf, int k0) {
    async_copy_b128((unsigned)(uintptr_t)&sA[buf][c_row0][c_col0],
                    A + (size_t)(m_blk + c_row0) * lda + k0 + c_col0);
    async_copy_b128((unsigned)(uintptr_t)&sA[buf][c_row1][c_col1],
                    A + (size_t)(m_blk + c_row1) * lda + k0 + c_col1);
    async_copy_b128((unsigned)(uintptr_t)&sB[buf][c_row0][c_col0],
                    W + (size_t)(n_blk + c_row0) * ldw + k0 + c_col0);
    async_copy_b128((unsigned)(uintptr_t)&sB[buf][c_row1][c_col1],
                    W + (size_t)(n_blk + c_row1) * ldw + k0 + c_col1);
  };

  const v8f z = {0.f, 0.f, 0.f, 0.f, 0.f, 0.f, 0.f, 0.f};
  v8f acc[2][4];
#pragma unroll
  for (int i = 0; i < 2; ++i)
#pragma unroll
    for (int j = 0; j < 4; ++j) acc[i][j] = z;

  const int nkt = K >> 5;

  // prologue: stage K-tile 0
  stage(0, 0);
  wait_asynccnt0();
  __syncthreads();

  for (int kt = 0; kt < nkt; ++kt) {
    const int cur = kt & 1;
    if (kt + 1 < nkt) stage(cur ^ 1, (kt + 1) * 32);

    // fragments from LDS
    v16bf a[2], b[4];
#pragma unroll
    for (int ms = 0; ms < 2; ++ms) {
      const unsigned short* ap = &sA[cur][wm * 32 + ms * 16 + llo][0];
      a[ms] = load_a_frag(ap, lhi);
    }
#pragma unroll
    for (int ns = 0; ns < 4; ++ns)
      b[ns] = load_b_frag(&sB[cur][wn * 64 + ns * 16 + llo][16 * lhi]);

#pragma unroll
    for (int ms = 0; ms < 2; ++ms)
#pragma unroll
      for (int ns = 0; ns < 4; ++ns)
        acc[ms][ns] = __builtin_amdgcn_wmma_f32_16x16x32_bf16(
            false, a[ms], false, b[ns], (short)0, acc[ms][ns], false, false);

    wait_asynccnt0();   // own async copies for next tile landed
    __syncthreads();    // cross-wave: copies visible, reads of 'cur' done
  }

#pragma unroll
  for (int ms = 0; ms < 2; ++ms)
#pragma unroll
    for (int ns = 0; ns < 4; ++ns)
#pragma unroll
      for (int r = 0; r < 8; ++r) {
        int row = m_blk + wm * 32 + ms * 16 + r + 8 * lhi;
        int col = n_blk + wn * 64 + ns * 16 + llo;
        C[(size_t)row * ldc + col] = acc[ms][ns][r];
      }
}

// ---------------- fused flash attention ----------------
// grid.x = B*H (64), grid.y = S/128 (16); 8 waves, each owns 16 query rows.
__global__ __launch_bounds__(256)
void flash_attn_kernel(const unsigned short* __restrict__ Qb,   // (BS, D) bf16, roped
                       const unsigned short* __restrict__ Kb,   // (BS, 512) bf16, roped
                       const unsigned short* __restrict__ Vt,   // (B*G*64, S) bf16
                       unsigned short* __restrict__ Yb) {       // (BS, D) bf16
  __shared__ unsigned short plds[8][16][32];  // per-wave P tile (16 q x 32 keys)

  const int lane = threadIdx.x & 31;
  const int wv   = threadIdx.x >> 5;
  const int lhi  = lane >> 4;
  const int llo  = lane & 15;
  const int bh = blockIdx.x;
  const int b  = bh >> 5;          // / NHEAD
  const int h  = bh & (NHEAD - 1);
  const int g  = h >> 2;           // / (H/G)
  const int q0 = blockIdx.y * 128 + wv * 16;

  // Q fragments: 16 rows x 64 dims = two K=32 A-fragments, kept in registers.
  v16bf qa[2];
  {
    const unsigned short* qp =
        Qb + (size_t)(b * S_LEN + q0 + llo) * DMODEL + h * HDIM;
    qa[0] = load_a_frag(qp, lhi);
    qa[1] = load_a_frag(qp + 32, lhi);
  }

  const v8f z = {0.f, 0.f, 0.f, 0.f, 0.f, 0.f, 0.f, 0.f};
  v8f o[4];
#pragma unroll
  for (int i = 0; i < 4; ++i) o[i] = z;
  float mrow[8], lrow[8];
#pragma unroll
  for (int r = 0; r < 8; ++r) { mrow[r] = -1e30f; lrow[r] = 0.f; }

  const float SCL = 0.125f * 1.44269504088896f;  // 1/sqrt(64) * log2(e)

  for (int t0 = 0; t0 < S_LEN; t0 += 32) {
    // --- scores: two 16x16 tiles over 32 keys ---
    v8f sc[2] = {z, z};
#pragma unroll
    for (int tile = 0; tile < 2; ++tile) {
      const unsigned short* kp =
          Kb + (size_t)(b * S_LEN + t0 + tile * 16 + llo) * KV_D + g * HDIM +
          16 * lhi;
      v16bf kb0 = load_b_frag(kp);        // d = 0..31
      v16bf kb1 = load_b_frag(kp + 32);   // d = 32..63
      sc[tile] = __builtin_amdgcn_wmma_f32_16x16x32_bf16(
          false, qa[0], false, kb0, (short)0, sc[tile], false, false);
      sc[tile] = __builtin_amdgcn_wmma_f32_16x16x32_bf16(
          false, qa[1], false, kb1, (short)0, sc[tile], false, false);
    }

    // --- online softmax (row = r + 8*lhi; reduce over the 16-lane N group) ---
    float alpha[8];
#pragma unroll
    for (int r = 0; r < 8; ++r) {
      float v0 = sc[0][r] * SCL;
      float v1 = sc[1][r] * SCL;
      float mx = fmaxf(v0, v1);
      mx = fmaxf(mx, __shfl_xor(mx, 1, 32));
      mx = fmaxf(mx, __shfl_xor(mx, 2, 32));
      mx = fmaxf(mx, __shfl_xor(mx, 4, 32));
      mx = fmaxf(mx, __shfl_xor(mx, 8, 32));
      float mn = fmaxf(mrow[r], mx);
      alpha[r] = __builtin_exp2f(mrow[r] - mn);
      mrow[r]  = mn;
      float p0 = __builtin_exp2f(v0 - mn);
      float p1 = __builtin_exp2f(v1 - mn);
      float ps = p0 + p1;
      ps += __shfl_xor(ps, 1, 32);
      ps += __shfl_xor(ps, 2, 32);
      ps += __shfl_xor(ps, 4, 32);
      ps += __shfl_xor(ps, 8, 32);
      lrow[r] = lrow[r] * alpha[r] + ps;
      // stage P (C-layout) into per-wave LDS as bf16
      plds[wv][r + 8 * lhi][llo]      = f32_to_bf16(p0);
      plds[wv][r + 8 * lhi][16 + llo] = f32_to_bf16(p1);
    }
    __builtin_amdgcn_wave_barrier();

    // --- re-read P in A-fragment layout (per-wave region, intra-wave dep) ---
    v16bf pa;
    ((uint4*)&pa)[0] = *(const uint4*)&plds[wv][llo][16 * lhi];
    ((uint4*)&pa)[1] = *(const uint4*)&plds[wv][llo][16 * lhi + 8];

    // --- O = O*alpha + P(16x32) x V(32x64) ---
#pragma unroll
    for (int ns = 0; ns < 4; ++ns) {
#pragma unroll
      for (int r = 0; r < 8; ++r) o[ns][r] *= alpha[r];
      const unsigned short* vp =
          Vt + ((size_t)(b * NKV + g) * HDIM + ns * 16 + llo) * S_LEN + t0 +
          16 * lhi;
      v16bf vb = load_b_frag(vp);
      o[ns] = __builtin_amdgcn_wmma_f32_16x16x32_bf16(
          false, pa, false, vb, (short)0, o[ns], false, false);
    }
  }

  // --- epilogue: normalize, write bf16 into Y at (b,s, h*64 + d) ---
#pragma unroll
  for (int ns = 0; ns < 4; ++ns)
#pragma unroll
    for (int r = 0; r < 8; ++r) {
      float val = o[ns][r] / lrow[r];
      size_t row = (size_t)(b * S_LEN + q0 + r + 8 * lhi);
      Yb[row * DMODEL + h * HDIM + ns * 16 + llo] = f32_to_bf16(val);
    }
}

// ---------------- host launch ----------------
extern "C" void kernel_launch(void* const* d_in, const int* in_sizes, int n_in,
                              void* d_out, int out_size, void* d_ws, size_t ws_size,
                              hipStream_t stream) {
  (void)in_sizes; (void)n_in; (void)out_size; (void)ws_size;
  const float* x    = (const float*)d_in[0];
  const float* cosp = (const float*)d_in[1];
  const float* sinp = (const float*)d_in[2];
  // d_in[3] = mask: intentionally unused (reference discards masked_fill result)
  const float* Wq = (const float*)d_in[4];
  const float* Wk = (const float*)d_in[5];
  const float* Wv = (const float*)d_in[6];
  const float* Wo = (const float*)d_in[7];
  float* out = (float*)d_out;

  char* ws = (char*)d_ws;
  size_t off = 0;
  auto alloc = [&](size_t bytes) -> void* {
    void* p = ws + off;
    off += (bytes + 255) & ~(size_t)255;
    return p;
  };
  unsigned short* xb  = (unsigned short*)alloc((size_t)BS * DMODEL * 2);
  unsigned short* Wqb = (unsigned short*)alloc((size_t)DMODEL * DMODEL * 2);
  unsigned short* Wkb = (unsigned short*)alloc((size_t)KV_D * DMODEL * 2);
  unsigned short* Wvb = (unsigned short*)alloc((size_t)KV_D * DMODEL * 2);
  unsigned short* Wob = (unsigned short*)alloc((size_t)DMODEL * DMODEL * 2);
  float* Qf = (float*)alloc((size_t)BS * DMODEL * 4);
  float* Kf = (float*)alloc((size_t)BS * KV_D * 4);
  float* Vf = (float*)alloc((size_t)BS * KV_D * 4);
  unsigned short* Qb  = (unsigned short*)alloc((size_t)BS * DMODEL * 2);
  unsigned short* Kb  = (unsigned short*)alloc((size_t)BS * KV_D * 2);
  unsigned short* Vtb = (unsigned short*)alloc((size_t)BATCH * NKV * HDIM * S_LEN * 2);
  unsigned short* Yb  = (unsigned short*)alloc((size_t)BS * DMODEL * 2);

  auto cvt = [&](const float* src, unsigned short* dst, size_t n) {
    cvt_bf16_kernel<<<dim3((unsigned)((n + 255) / 256)), 256, 0, stream>>>(src, dst, n);
  };
  cvt(x,  xb,  (size_t)BS * DMODEL);
  cvt(Wq, Wqb, (size_t)DMODEL * DMODEL);
  cvt(Wk, Wkb, (size_t)KV_D * DMODEL);
  cvt(Wv, Wvb, (size_t)KV_D * DMODEL);
  cvt(Wo, Wob, (size_t)DMODEL * DMODEL);

  // projections: y = x @ W^T
  gemm_bf16_nt<<<dim3(DMODEL / 128, BS / 128), 256, 0, stream>>>(
      BS, DMODEL, DMODEL, xb, DMODEL, Wqb, DMODEL, Qf, DMODEL);
  gemm_bf16_nt<<<dim3(KV_D / 128, BS / 128), 256, 0, stream>>>(
      BS, KV_D, DMODEL, xb, DMODEL, Wkb, DMODEL, Kf, KV_D);
  gemm_bf16_nt<<<dim3(KV_D / 128, BS / 128), 256, 0, stream>>>(
      BS, KV_D, DMODEL, xb, DMODEL, Wvb, DMODEL, Vf, KV_D);

  // RoPE + bf16 convert
  {
    size_t nq = (size_t)BS * DMODEL;
    rope_cvt_kernel<<<dim3((unsigned)((nq + 255) / 256)), 256, 0, stream>>>(
        Qf, cosp, sinp, Qb, DMODEL);
    size_t nk = (size_t)BS * KV_D;
    rope_cvt_kernel<<<dim3((unsigned)((nk + 255) / 256)), 256, 0, stream>>>(
        Kf, cosp, sinp, Kb, KV_D);
    vt_cvt_kernel<<<dim3((unsigned)((nk + 255) / 256)), 256, 0, stream>>>(Vf, Vtb);
  }

  // fused attention
  flash_attn_kernel<<<dim3(BATCH * NHEAD, S_LEN / 128), 256, 0, stream>>>(
      Qb, Kb, Vtb, Yb);

  // output projection -> fp32 result
  gemm_bf16_nt<<<dim3(DMODEL / 128, BS / 128), 256, 0, stream>>>(
      BS, DMODEL, DMODEL, Yb, DMODEL, Wob, DMODEL, out, DMODEL);
}
